// Discriminator_87900800680098
// MI455X (gfx1250) — matrix operand order repaired and checked by reading, not compiled, not run on hardware
//
#include <hip/hip_runtime.h>
#include <hip/hip_bf16.h>

typedef __attribute__((ext_vector_type(16))) __bf16 v16bf;
typedef __attribute__((ext_vector_type(8)))  __bf16 v8bf;
typedef __attribute__((ext_vector_type(8)))  float  v8f;

#define H_DIM 512
#define G_DIM 2048
#define B_DIM 256
#define T_DIM 256
#define I_DIM 128

// ---- workspace layout (byte offsets, all 32B-aligned) ----
// zero-init block: [c0, c1, h0hi0, h0lo0, h1hi0, h1lo0] = 2 MB contiguous
#define OFF_C0      0u          // 131072 f32
#define OFF_C1      524288u     // 131072 f32
#define OFF_H0HI0   1048576u    // 131072 bf16
#define OFF_H0LO0   1310720u
#define OFF_H1HI0   1572864u
#define OFF_H1LO0   1835008u
#define OFF_H0HI1   2097152u
#define OFF_H0LO1   2359296u
#define OFF_H1HI1   2621440u
#define OFF_H1LO1   2883584u
#define OFF_BIAS0   3145728u    // 2048 f32
#define OFF_BIAS1   3153920u
#define OFF_XHI     3162112u    // 8388608 bf16
#define OFF_XLO     19939328u
#define OFF_WIH0HI  36716544u   // 262144 bf16
#define OFF_WIH0LO  37240832u
#define OFF_WHH0HI  37765120u   // 1048576 bf16
#define OFF_WHH0LO  39862272u
#define OFF_WIH1HI  41959424u
#define OFF_WIH1LO  44056576u
#define OFF_WHH1HI  46153728u
#define OFF_WHH1LO  48250880u

__global__ void zero_kernel(unsigned int* p, int n) {
  int i = blockIdx.x * blockDim.x + threadIdx.x;
  if (i < n) p[i] = 0u;
}

__global__ void bias_kernel(const float* __restrict__ bi0, const float* __restrict__ bh0,
                            const float* __restrict__ bi1, const float* __restrict__ bh1,
                            float* __restrict__ bias0, float* __restrict__ bias1) {
  int i = blockIdx.x * blockDim.x + threadIdx.x;
  if (i < G_DIM) {
    bias0[i] = bi0[i] + bh0[i];
    bias1[i] = bi1[i] + bh1[i];
  }
}

// fp32 -> split bf16 (hi + residual lo); run once per launch for weights and x
__global__ void cvt_kernel(const float* __restrict__ src, __bf16* __restrict__ hi,
                           __bf16* __restrict__ lo, int n) {
  int i = blockIdx.x * blockDim.x + threadIdx.x;
  if (i < n) {
    float f = src[i];
    __bf16 h = (__bf16)f;
    hi[i] = h;
    lo[i] = (__bf16)(f - (float)h);
  }
}

__device__ __forceinline__ float sigmoidf_(float x) { return 1.0f / (1.0f + __expf(-x)); }
__device__ __forceinline__ float tanhf_(float x)    { return 1.0f - 2.0f / (__expf(2.0f * x) + 1.0f); }

// Load one split-bf16 A operand fragment (16x32 A, this code's own k order, not the ISA table):
// lanes 0-15: K = k+0..15 ; lanes 16-31: K = k+16..31 (contiguous, matches B)
__device__ __forceinline__ v16bf load_a(const __bf16* row, int kbase) {
  v8bf q0 = *(const v8bf*)(row + kbase);
  v8bf q1 = *(const v8bf*)(row + kbase + 8);
  return __builtin_shufflevector(q0, q1, 0, 1, 2, 3, 4, 5, 6, 7,
                                         8, 9, 10, 11, 12, 13, 14, 15);
}

// One LSTM timestep of one layer. A-side operands are pre-split bf16; the
// hot loop is pure VMEM + WMMA. One wave owns a 32(M)x16(N-of-H) tile and
// all 4 gate accumulators (i,f,g,o share the A operand).
__global__ __launch_bounds__(32)
void lstm_step(const __bf16* __restrict__ a1_hi, const __bf16* __restrict__ a1_lo,
               int a1_stride, int k1,
               const __bf16* __restrict__ wih_hi, const __bf16* __restrict__ wih_lo,
               const __bf16* __restrict__ hp_hi, const __bf16* __restrict__ hp_lo,
               const __bf16* __restrict__ whh_hi, const __bf16* __restrict__ whh_lo,
               const float* __restrict__ bias,
               float* __restrict__ c_state,
               __bf16* __restrict__ ho_hi, __bf16* __restrict__ ho_lo)
{
  const int lane = threadIdx.x;
  const int lm   = lane & 15;
  const int half = lane >> 4;
  const int m0   = blockIdx.x * 32;   // two 16-row M subtiles
  const int n0   = blockIdx.y * 16;   // 16 hidden columns

  v8f acc[2][4];
#pragma unroll
  for (int s = 0; s < 2; ++s)
#pragma unroll
    for (int g = 0; g < 4; ++g)
      acc[s][g] = (v8f){0.f, 0.f, 0.f, 0.f, 0.f, 0.f, 0.f, 0.f};

  for (int phase = 0; phase < 2; ++phase) {
    const __bf16* shi    = phase ? hp_hi  : a1_hi;
    const __bf16* slo    = phase ? hp_lo  : a1_lo;
    const int     stride = phase ? H_DIM  : a1_stride;
    const int     K      = phase ? H_DIM  : k1;
    const __bf16* whi    = phase ? whh_hi : wih_hi;
    const __bf16* wlo    = phase ? whh_lo : wih_lo;

    for (int k0 = 0; k0 < K; k0 += 32) {
      v16bf ahi[2], alo[2];
#pragma unroll
      for (int s = 0; s < 2; ++s) {
        size_t roff = (size_t)(m0 + s * 16 + lm) * stride;
        int kb = k0 + half * 16;
        ahi[s] = load_a(shi + roff, kb);
        alo[s] = load_a(slo + roff, kb);
      }
      // B (32x16, this code's own k order, not the ISA table): lane half selects K = k0+0..15 / k0+16..31, contiguous
#pragma unroll
      for (int g = 0; g < 4; ++g) {
        const size_t wrow = (size_t)(g * H_DIM + n0 + lm) * (size_t)K;
        v16bf bh = *(const v16bf*)(whi + wrow + k0 + half * 16);
        v16bf bl = *(const v16bf*)(wlo + wrow + k0 + half * 16);
#pragma unroll
        for (int s = 0; s < 2; ++s) {
          acc[s][g] = __builtin_amdgcn_wmma_f32_16x16x32_bf16(false, ahi[s], false, bh,
                                                              (short)0, acc[s][g], false, false);
          acc[s][g] = __builtin_amdgcn_wmma_f32_16x16x32_bf16(false, ahi[s], false, bl,
                                                              (short)0, acc[s][g], false, false);
          acc[s][g] = __builtin_amdgcn_wmma_f32_16x16x32_bf16(false, alo[s], false, bh,
                                                              (short)0, acc[s][g], false, false);
        }
      }
    }
  }

  // bias broadcast: C/D layout has N = lane%16 for every accumulator VGPR
  float bv[4];
#pragma unroll
  for (int g = 0; g < 4; ++g) bv[g] = bias[g * H_DIM + n0 + lm];

  const int ncol = n0 + lm;
#pragma unroll
  for (int s = 0; s < 2; ++s) {
#pragma unroll
    for (int r = 0; r < 8; ++r) {
      int m = m0 + s * 16 + r + half * 8;       // C/D layout: M = vgpr + 8*(lane>=16)
      float gi = acc[s][0][r] + bv[0];
      float gf = acc[s][1][r] + bv[1];
      float gg = acc[s][2][r] + bv[2];
      float go = acc[s][3][r] + bv[3];
      float iv = sigmoidf_(gi);
      float fv = sigmoidf_(gf);
      float gv = tanhf_(gg);
      float ov = sigmoidf_(go);
      size_t idx = (size_t)m * H_DIM + ncol;
      float cnew = fv * c_state[idx] + iv * gv;
      c_state[idx] = cnew;
      float hv = ov * tanhf_(cnew);
      __bf16 hh = (__bf16)hv;                   // split for next step's A operand
      ho_hi[idx] = hh;
      ho_lo[idx] = (__bf16)(hv - (float)hh);
    }
  }
}

__global__ void fc_kernel(const __bf16* __restrict__ h_hi, const __bf16* __restrict__ h_lo,
                          const float* __restrict__ w, const float* __restrict__ b,
                          float* __restrict__ out) {
  int i = threadIdx.x;  // 256 threads, one batch row each
  float s = b[0];
  const __bf16* hh = h_hi + (size_t)i * H_DIM;
  const __bf16* hl = h_lo + (size_t)i * H_DIM;
#pragma unroll 8
  for (int k = 0; k < H_DIM; ++k) s += ((float)hh[k] + (float)hl[k]) * w[k];
  out[i] = 1.0f / (1.0f + __expf(-s));
}

extern "C" void kernel_launch(void* const* d_in, const int* in_sizes, int n_in,
                              void* d_out, int out_size, void* d_ws, size_t ws_size,
                              hipStream_t stream) {
  const float* x    = (const float*)d_in[0];
  const float* wih0 = (const float*)d_in[1];
  const float* whh0 = (const float*)d_in[2];
  const float* bih0 = (const float*)d_in[3];
  const float* bhh0 = (const float*)d_in[4];
  const float* wih1 = (const float*)d_in[5];
  const float* whh1 = (const float*)d_in[6];
  const float* bih1 = (const float*)d_in[7];
  const float* bhh1 = (const float*)d_in[8];
  const float* wfc  = (const float*)d_in[9];
  const float* bfc  = (const float*)d_in[10];

  char* ws = (char*)d_ws;
  float*  c0    = (float*)(ws + OFF_C0);
  float*  c1    = (float*)(ws + OFF_C1);
  float*  bias0 = (float*)(ws + OFF_BIAS0);
  float*  bias1 = (float*)(ws + OFF_BIAS1);
  __bf16* h0hi[2] = { (__bf16*)(ws + OFF_H0HI0), (__bf16*)(ws + OFF_H0HI1) };
  __bf16* h0lo[2] = { (__bf16*)(ws + OFF_H0LO0), (__bf16*)(ws + OFF_H0LO1) };
  __bf16* h1hi[2] = { (__bf16*)(ws + OFF_H1HI0), (__bf16*)(ws + OFF_H1HI1) };
  __bf16* h1lo[2] = { (__bf16*)(ws + OFF_H1LO0), (__bf16*)(ws + OFF_H1LO1) };
  __bf16* x_hi    = (__bf16*)(ws + OFF_XHI);
  __bf16* x_lo    = (__bf16*)(ws + OFF_XLO);
  __bf16* wih0_hi = (__bf16*)(ws + OFF_WIH0HI);
  __bf16* wih0_lo = (__bf16*)(ws + OFF_WIH0LO);
  __bf16* whh0_hi = (__bf16*)(ws + OFF_WHH0HI);
  __bf16* whh0_lo = (__bf16*)(ws + OFF_WHH0LO);
  __bf16* wih1_hi = (__bf16*)(ws + OFF_WIH1HI);
  __bf16* wih1_lo = (__bf16*)(ws + OFF_WIH1LO);
  __bf16* whh1_hi = (__bf16*)(ws + OFF_WHH1HI);
  __bf16* whh1_lo = (__bf16*)(ws + OFF_WHH1LO);

  // zero c0,c1 + t=0 h ping buffers (2 MB contiguous)
  zero_kernel<<<524288 / 256, 256, 0, stream>>>((unsigned int*)ws, 524288);
  bias_kernel<<<8, 256, 0, stream>>>(bih0, bhh0, bih1, bhh1, bias0, bias1);
  // one-time split-bf16 conversion: input tensor + all four weight matrices
  cvt_kernel<<<8388608 / 256, 256, 0, stream>>>(x, x_hi, x_lo, 8388608);
  cvt_kernel<<<262144 / 256, 256, 0, stream>>>(wih0, wih0_hi, wih0_lo, 262144);
  cvt_kernel<<<1048576 / 256, 256, 0, stream>>>(whh0, whh0_hi, whh0_lo, 1048576);
  cvt_kernel<<<1048576 / 256, 256, 0, stream>>>(wih1, wih1_hi, wih1_lo, 1048576);
  cvt_kernel<<<1048576 / 256, 256, 0, stream>>>(whh1, whh1_hi, whh1_lo, 1048576);

  dim3 grid(B_DIM / 32, H_DIM / 16), blk(32);
  for (int t = 0; t < T_DIM; ++t) {
    int pi = t & 1, po = (t + 1) & 1;
    // layer 0: A1 = x[:, t, :]  (row stride T*I, pre-split bf16)
    lstm_step<<<grid, blk, 0, stream>>>(x_hi + (size_t)t * I_DIM, x_lo + (size_t)t * I_DIM,
                                        T_DIM * I_DIM, I_DIM,
                                        wih0_hi, wih0_lo,
                                        h0hi[pi], h0lo[pi], whh0_hi, whh0_lo,
                                        bias0, c0, h0hi[po], h0lo[po]);
    // layer 1: A1 = h0 just produced
    lstm_step<<<grid, blk, 0, stream>>>(h0hi[po], h0lo[po], H_DIM, H_DIM,
                                        wih1_hi, wih1_lo,
                                        h1hi[pi], h1lo[pi], whh1_hi, whh1_lo,
                                        bias1, c1, h1hi[po], h1lo[po]);
  }
  // after t=255, layer-1 output lives in buffer 0
  fc_kernel<<<1, 256, 0, stream>>>(h1hi[0], h1lo[0], wfc, bfc, (float*)d_out);
}
